// HierLstmat_64166811402286
// MI455X (gfx1250) — matrix operand order repaired and checked by reading, not compiled, not run on hardware
//
#include <hip/hip_runtime.h>

// ---------------------------------------------------------------------------
// HierLstmat on MI455X (gfx1250): bf16 WMMA for all GEMMs, f32 accumulation.
// bf16 storage type for host-visible pointers (avoid __bf16 in host decls).
// ---------------------------------------------------------------------------
typedef unsigned short bf16_t;

typedef __attribute__((ext_vector_type(16))) __bf16 v16bf;
typedef __attribute__((ext_vector_type(8)))  __bf16 v8bf;
typedef __attribute__((ext_vector_type(8)))  float  v8f;

#define Bsz 32
#define Tsz 6400
#define Csz 256
#define Hsz 256
#define UPL 80          // chunk length
#define NCH 80          // number of chunks
#define G4H 1024        // 4*H
#define ROWS (UPL*Bsz)  // 2560 rows per chunk GEMM

__device__ __forceinline__ float sigmoidf_(float x) { return 1.0f / (1.0f + __expf(-x)); }

__device__ __forceinline__ v8f wmma_bf16(v16bf a, v16bf b, v8f c) {
    return __builtin_amdgcn_wmma_f32_16x16x32_bf16(false, a, false, b, (short)0, c, false, false);
}

// A fragment: 16x32 tile from row-major bf16 [rows][ld], tile origin (row0,k0).
// ISA 7.12.2: lanes 0-15 -> M=lane, K = k0+{0..7,16..23}; lanes 16-31 -> K = k0+{8..15,24..31}.
__device__ __forceinline__ v16bf load_frag_a(const __bf16* base, int ld, int row0, int k0, int lane) {
    int r  = row0 + (lane & 15);
    int kb = k0 + ((lane & 16) ? 8 : 0);
    const __bf16* p = base + r * ld + kb;
    v8bf lo = *(const v8bf*)(p);
    v8bf hi = *(const v8bf*)(p + 16);
    v16bf a;
#pragma unroll
    for (int i = 0; i < 8; ++i) { a[i] = lo[i]; a[i + 8] = hi[i]; }
    return a;
}

// B fragment (32x16, K x N) sourced from weight W stored row-major [N][K]
// (B[k][n] = W[n][k]). Lane n'=lane&15 -> col n0+n'; lanes 0-15 K = k0+{0..7,16..23}; lanes 16-31 K = k0+{8..15,24..31}.
__device__ __forceinline__ v16bf load_frag_b(const __bf16* W, int ld, int n0, int k0, int lane) {
    const __bf16* p = W + (n0 + (lane & 15)) * ld + k0 + ((lane & 16) ? 8 : 0);
    v8bf lo = *(const v8bf*)(p);
    v8bf hi = *(const v8bf*)(p + 16);
    v16bf b;
#pragma unroll
    for (int i = 0; i < 8; ++i) { b[i] = lo[i]; b[i + 8] = hi[i]; }
    return b;
}

// ---------------------------------------------------------------------------
// Prep kernels
// ---------------------------------------------------------------------------
__global__ void k_cvt(bf16_t* dst_, const float* __restrict__ src, int n) {
    int i = blockIdx.x * blockDim.x + threadIdx.x;
    if (i < n) ((__bf16*)dst_)[i] = (__bf16)src[i];
}

__global__ void k_wcat2(bf16_t* dst_, const float* __restrict__ wih2, const float* __restrict__ whh2) {
    int i = blockIdx.x * blockDim.x + threadIdx.x;
    if (i < G4H * 512) {
        int nr = i >> 9, k = i & 511;
        ((__bf16*)dst_)[i] = (__bf16)(k < 256 ? wih2[nr * 256 + k] : whh2[nr * 256 + (k - 256)]);
    }
}

__global__ void k_bias_sum(float* dst, const float* __restrict__ a, const float* __restrict__ b, int n) {
    int i = blockIdx.x * blockDim.x + threadIdx.x;
    if (i < n) dst[i] = a[i] + b[i];
}

__global__ void k_init_state(float* st, const float* __restrict__ h0, const float* __restrict__ c0,
                             const float* __restrict__ h20, const float* __restrict__ c20) {
    int i = blockIdx.x * blockDim.x + threadIdx.x;
    if (i < Bsz * Hsz) {
        st[i]                 = h0[i];
        st[Bsz * Hsz     + i] = c0[i];
        st[2 * Bsz * Hsz + i] = h20[i];
        st[3 * Bsz * Hsz + i] = c20[i];
    }
}

// x chunk [B,T,C] f32 -> xh [(l*32+b)][k] bf16
__global__ void k_cvt_x(bf16_t* xh_, const float* __restrict__ x, int t0) {
    int i = blockIdx.x * blockDim.x + threadIdx.x;
    if (i < ROWS * Csz) {
        int l = i >> 13;          // /(32*256)
        int b = (i >> 8) & 31;
        int k = i & 255;
        ((__bf16*)xh_)[i] = (__bf16)x[(b * Tsz + t0 + l) * Csz + k];
    }
}

// ---------------------------------------------------------------------------
// XG = xh @ W_ih1^T + (b_ih1+b_hh1).  M=2560, N=1024, K=256.
// 10240 tiles -> 1280 blocks x 256 threads (8 waves), one 16x16 tile per wave.
// ---------------------------------------------------------------------------
__global__ void k_xgate(const bf16_t* __restrict__ xh_, const bf16_t* __restrict__ Wih1_,
                        const float* __restrict__ bias1, float* __restrict__ XG) {
    const __bf16* xh = (const __bf16*)xh_;
    const __bf16* W  = (const __bf16*)Wih1_;
    int wave = (blockIdx.x * blockDim.x + threadIdx.x) >> 5;
    int lane = threadIdx.x & 31;
    int mt = wave >> 6, nt = wave & 63;
    v8f acc = {};
#pragma unroll
    for (int kb = 0; kb < 8; ++kb) {
        v16bf a = load_frag_a(xh, Csz, mt * 16, kb * 32, lane);
        v16bf b = load_frag_b(W,  Csz, nt * 16, kb * 32, lane);
        acc = wmma_bf16(a, b, acc);
    }
    int rrow = (lane & 16) ? 8 : 0;
    int n = nt * 16 + (lane & 15);
#pragma unroll
    for (int r = 0; r < 8; ++r) {
        int m = mt * 16 + rrow + r;
        XG[m * G4H + n] = acc[r] + bias1[n];
    }
}

// ---------------------------------------------------------------------------
// Per-chunk sequential kernel: 80 LSTM-1 steps (WMMA h@Whh1^T), attention
// pooling of hs and cs(+c2), LSTM-2 cell (WMMA over K=512 concat).
// One workgroup: 1024 threads = 32 wave32s; ~260KB dynamic LDS (CDNA5: 320KB/WGP).
// ---------------------------------------------------------------------------
#define CS_LDS_BYTES 266368

__global__ void __launch_bounds__(1024)
k_chunk_seq(const float* __restrict__ XG,
            const bf16_t* __restrict__ Whh1_,
            const bf16_t* __restrict__ Wcat2_,
            const float* __restrict__ bias2,
            const float* __restrict__ Wp, const float* __restrict__ bp,
            float* state,            // h1[8192] c1[8192] h2[8192] c2[8192]
            float* __restrict__ hs,  // [80][32][256]
            float* __restrict__ cs,  // [80][32][256]
            float* __restrict__ outerh,     // [80][32][256]
            bf16_t* __restrict__ outerhb_,  // bf16 copy
            int chunk)
{
    extern __shared__ char smem[];
    float*  gL  = (float*)smem;          // 32*1024 gates
    float*  cL  = gL  + 32 * 1024;       // 32*256  c1
    float*  c2p = cL  + 32 * 256;        // 32*256  pooled c2
    float*  sm1 = c2p + 32 * 256;        // 32*80   softmax(hs)
    float*  sm2 = sm1 + 32 * 80;         // 32*81   softmax(cs_aug)
    __bf16* hA  = (__bf16*)(sm2 + 32 * 81);   // 32*256 A-frag for layer1
    __bf16* hA2 = hA + 32 * 256;              // 32*512 A-frag for layer2

    const __bf16* Whh1  = (const __bf16*)Whh1_;
    const __bf16* Wcat2 = (const __bf16*)Wcat2_;
    __bf16* outerhb = (__bf16*)outerhb_;

    int tid  = threadIdx.x;
    int lane = tid & 31;
    int wave = tid >> 5;
    int mt   = wave >> 4;            // 2 M-tiles across 32 waves
    int ntb  = (wave & 15) * 4;      // 4 N-tiles per wave
    int rrow = (lane & 16) ? 8 : 0;
    int ncol = lane & 15;

    // load carried state
    for (int idx = tid; idx < Bsz * Hsz; idx += 1024) {
        hA[idx] = (__bf16)state[idx];          // h1, row-major [b][j]
        cL[idx] = state[Bsz * Hsz + idx];      // c1
    }
    __syncthreads();

    // ---- 80 sequential LSTM-1 steps ----
    for (int l = 0; l < UPL; ++l) {
        v8f acc[4] = {};
#pragma unroll
        for (int kb = 0; kb < 8; ++kb) {
            v16bf a = load_frag_a(hA, Hsz, mt * 16, kb * 32, lane);
#pragma unroll
            for (int q = 0; q < 4; ++q) {
                v16bf b = load_frag_b(Whh1, Hsz, (ntb + q) * 16, kb * 32, lane);
                acc[q] = wmma_bf16(a, b, acc[q]);
            }
        }
        const float* xg = XG + (size_t)l * Bsz * G4H;
#pragma unroll
        for (int q = 0; q < 4; ++q) {
            int n0 = (ntb + q) * 16 + ncol;
#pragma unroll
            for (int r = 0; r < 8; ++r) {
                int m = mt * 16 + rrow + r;
                gL[m * G4H + n0] = acc[q][r] + xg[m * G4H + n0];
            }
        }
        __syncthreads();
        for (int idx = tid; idx < Bsz * Hsz; idx += 1024) {
            int b = idx >> 8, j = idx & 255;
            float gi = gL[b * G4H + j];
            float gf = gL[b * G4H + 256 + j];
            float gg = gL[b * G4H + 512 + j];
            float go = gL[b * G4H + 768 + j];
            float c = sigmoidf_(gf) * cL[idx] + sigmoidf_(gi) * tanhf(gg);
            float h = sigmoidf_(go) * tanhf(c);
            cL[idx] = c;
            hA[idx] = (__bf16)h;
            hs[(l * Bsz + b) * Hsz + j] = h;
            cs[(l * Bsz + b) * Hsz + j] = c;
        }
        __syncthreads();
    }
    __threadfence();   // make hs/cs global writes visible to this block's reads
    __syncthreads();

    // ---- attention-pool scores ----
    for (int idx = tid; idx < Bsz * UPL; idx += 1024) {
        int b = idx / UPL, l = idx % UPL;
        const float* hr = hs + (l * Bsz + b) * Hsz;
        const float* cr = cs + (l * Bsz + b) * Hsz;
        float s1 = 0.f, s2 = 0.f;
        for (int j = 0; j < Hsz; ++j) { s1 += hr[j] * Wp[j]; s2 += cr[j] * Wp[j]; }
        sm1[b * 80 + l] = s1 + bp[0];
        sm2[b * 81 + l] = s2 + bp[0];
    }
    if (tid < Bsz) {  // appended c2 row for cs_aug
        const float* cr = state + 3 * Bsz * Hsz + tid * Hsz;
        float s2 = 0.f;
        for (int j = 0; j < Hsz; ++j) s2 += cr[j] * Wp[j];
        sm2[tid * 81 + 80] = s2 + bp[0];
    }
    __syncthreads();
    if (tid < 32) {
        float* s = sm1 + tid * 80;
        float mx = -1e30f;
        for (int i = 0; i < 80; ++i) mx = fmaxf(mx, s[i]);
        float sum = 0.f;
        for (int i = 0; i < 80; ++i) { float e = __expf(s[i] - mx); s[i] = e; sum += e; }
        float inv = 1.f / sum;
        for (int i = 0; i < 80; ++i) s[i] *= inv;
    } else if (tid < 64) {
        float* s = sm2 + (tid - 32) * 81;
        float mx = -1e30f;
        for (int i = 0; i < 81; ++i) mx = fmaxf(mx, s[i]);
        float sum = 0.f;
        for (int i = 0; i < 81; ++i) { float e = __expf(s[i] - mx); s[i] = e; sum += e; }
        float inv = 1.f / sum;
        for (int i = 0; i < 81; ++i) s[i] *= inv;
    }
    __syncthreads();

    // ---- pooled vectors + layer-2 A fragment [l2in | h2] ----
    for (int idx = tid; idx < Bsz * Hsz; idx += 1024) {
        int b = idx >> 8, j = idx & 255;
        float p1 = 0.f, p2 = 0.f;
        for (int l2 = 0; l2 < UPL; ++l2) {
            p1 += hs[(l2 * Bsz + b) * Hsz + j] * sm1[b * 80 + l2];
            p2 += cs[(l2 * Bsz + b) * Hsz + j] * sm2[b * 81 + l2];
        }
        p2 += state[3 * Bsz * Hsz + idx] * sm2[b * 81 + 80];
        c2p[idx] = p2;
        hA2[b * 512 + j]       = (__bf16)p1;
        hA2[b * 512 + 256 + j] = (__bf16)state[2 * Bsz * Hsz + idx];
    }
    __syncthreads();

    // ---- layer-2 gates: [32,512] x [512,1024] ----
    {
        v8f acc[4] = {};
#pragma unroll
        for (int kb = 0; kb < 16; ++kb) {
            v16bf a = load_frag_a(hA2, 512, mt * 16, kb * 32, lane);
#pragma unroll
            for (int q = 0; q < 4; ++q) {
                v16bf b = load_frag_b(Wcat2, 512, (ntb + q) * 16, kb * 32, lane);
                acc[q] = wmma_bf16(a, b, acc[q]);
            }
        }
#pragma unroll
        for (int q = 0; q < 4; ++q) {
            int n0 = (ntb + q) * 16 + ncol;
#pragma unroll
            for (int r = 0; r < 8; ++r) {
                int m = mt * 16 + rrow + r;
                gL[m * G4H + n0] = acc[q][r] + bias2[n0];
            }
        }
    }
    __syncthreads();

    // ---- layer-2 cell update + carry states ----
    for (int idx = tid; idx < Bsz * Hsz; idx += 1024) {
        int b = idx >> 8, j = idx & 255;
        float gi = gL[b * G4H + j];
        float gf = gL[b * G4H + 256 + j];
        float gg = gL[b * G4H + 512 + j];
        float go = gL[b * G4H + 768 + j];
        float c2n = sigmoidf_(gf) * c2p[idx] + sigmoidf_(gi) * tanhf(gg);
        float h2n = sigmoidf_(go) * tanhf(c2n);
        state[2 * Bsz * Hsz + idx] = h2n;
        state[3 * Bsz * Hsz + idx] = c2n;
        outerh [(chunk * Bsz + b) * Hsz + j] = h2n;
        outerhb[(chunk * Bsz + b) * Hsz + j] = 0;  // placeholder overwritten below
        ((__bf16*)outerhb)[(chunk * Bsz + b) * Hsz + j] = (__bf16)h2n;
        // layer-1 carry: h1 = hs[last], c1 from LDS
        state[idx]             = hs[(79 * Bsz + b) * Hsz + j];
        state[Bsz * Hsz + idx] = cL[idx];
    }
}

// ---------------------------------------------------------------------------
// ut = tanh(outerh @ W_u^T + b_u): M=2560, N=128, K=256 -> 1280 tiles.
// ---------------------------------------------------------------------------
__global__ void k_enc_ut(const bf16_t* __restrict__ oh_, const bf16_t* __restrict__ Wu_,
                         const float* __restrict__ bu, float* __restrict__ ut) {
    const __bf16* A = (const __bf16*)oh_;
    const __bf16* W = (const __bf16*)Wu_;
    int wave = (blockIdx.x * blockDim.x + threadIdx.x) >> 5;
    int lane = threadIdx.x & 31;
    int mt = wave >> 3, nt = wave & 7;
    v8f acc = {};
#pragma unroll
    for (int kb = 0; kb < 8; ++kb) {
        v16bf a = load_frag_a(A, Hsz, mt * 16, kb * 32, lane);
        v16bf b = load_frag_b(W, Hsz, nt * 16, kb * 32, lane);
        acc = wmma_bf16(a, b, acc);
    }
    int rrow = (lane & 16) ? 8 : 0;
    int n = nt * 16 + (lane & 15);
#pragma unroll
    for (int r = 0; r < 8; ++r)
        ut[(mt * 16 + rrow + r) * 128 + n] = tanhf(acc[r] + bu[n]);
}

// ---------------------------------------------------------------------------
// et -> softmax over chunks -> att_out -> flat (bf16).  One block.
// ---------------------------------------------------------------------------
__global__ void k_enc_att(const float* __restrict__ ut, const float* __restrict__ We,
                          const float* __restrict__ be, const float* __restrict__ outerh,
                          bf16_t* __restrict__ flat_) {
    __shared__ float sL[32 * 4 * 80];
    __bf16* flat = (__bf16*)flat_;
    int tid = threadIdx.x;
    for (int idx = tid; idx < 32 * 80 * 4; idx += 1024) {
        int b = idx / 320, rem = idx % 320, n = rem / 4, p = rem % 4;
        const float* u = ut + (n * Bsz + b) * 128;
        const float* w = We + p * 128;
        float e = be[p];
        for (int a = 0; a < 128; ++a) e += u[a] * w[a];
        sL[(b * 4 + p) * 80 + n] = e;
    }
    __syncthreads();
    if (tid < 128) {
        float* s = sL + tid * 80;
        float mx = -1e30f;
        for (int i = 0; i < 80; ++i) mx = fmaxf(mx, s[i]);
        float sum = 0.f;
        for (int i = 0; i < 80; ++i) { float e = __expf(s[i] - mx); s[i] = e; sum += e; }
        float inv = 1.f / sum;
        for (int i = 0; i < 80; ++i) s[i] *= inv;
    }
    __syncthreads();
    for (int idx = tid; idx < 32 * 4 * 256; idx += 1024) {
        int b = idx >> 10, p = (idx >> 8) & 3, h = idx & 255;
        const float* s = sL + (b * 4 + p) * 80;
        float v = 0.f;
        for (int n = 0; n < 80; ++n) v += s[n] * outerh[(n * Bsz + b) * Hsz + h];
        flat[b * 1024 + p * 256 + h] = (__bf16)v;
    }
}

// ---------------------------------------------------------------------------
// MLP head: hid = relu(flat@W_o1^T+b1) [32,512]; out = hid@W_o2^T+b2 [32,256].
// One block, 32 waves, both GEMMs with WMMA.
// ---------------------------------------------------------------------------
__global__ void __launch_bounds__(1024)
k_mlp(const bf16_t* __restrict__ flat_, const bf16_t* __restrict__ Wo1_,
      const float* __restrict__ bo1, const bf16_t* __restrict__ Wo2_,
      const float* __restrict__ bo2, float* __restrict__ out) {
    __shared__ __bf16 hidb[32 * 512];
    const __bf16* flat = (const __bf16*)flat_;
    const __bf16* Wo1  = (const __bf16*)Wo1_;
    const __bf16* Wo2  = (const __bf16*)Wo2_;
    int tid = threadIdx.x, lane = tid & 31, wave = tid >> 5;
    int rrow = (lane & 16) ? 8 : 0;
    int ncol = lane & 15;
    {   // phase 1: M=2,N=32 tiles; 2 tiles/wave; K=1024
        int mt = wave >> 4;
        int ntb = (wave & 15) * 2;
        v8f acc[2] = {};
#pragma unroll
        for (int kb = 0; kb < 32; ++kb) {
            v16bf a = load_frag_a(flat, 1024, mt * 16, kb * 32, lane);
#pragma unroll
            for (int q = 0; q < 2; ++q) {
                v16bf b = load_frag_b(Wo1, 1024, (ntb + q) * 16, kb * 32, lane);
                acc[q] = wmma_bf16(a, b, acc[q]);
            }
        }
#pragma unroll
        for (int q = 0; q < 2; ++q) {
            int n = (ntb + q) * 16 + ncol;
#pragma unroll
            for (int r = 0; r < 8; ++r) {
                int m = mt * 16 + rrow + r;
                hidb[m * 512 + n] = (__bf16)fmaxf(acc[q][r] + bo1[n], 0.f);
            }
        }
    }
    __syncthreads();
    {   // phase 2: M=2,N=16 tiles; 1 tile/wave; K=512
        int mt = wave >> 4, nt = wave & 15;
        v8f acc = {};
#pragma unroll
        for (int kb = 0; kb < 16; ++kb) {
            v16bf a = load_frag_a(hidb, 512, mt * 16, kb * 32, lane);
            v16bf b = load_frag_b(Wo2, 512, nt * 16, kb * 32, lane);
            acc = wmma_bf16(a, b, acc);
        }
        int n = nt * 16 + ncol;
#pragma unroll
        for (int r = 0; r < 8; ++r)
            out[(mt * 16 + rrow + r) * 256 + n] = acc[r] + bo2[n];
    }
}

// ---------------------------------------------------------------------------
// Host orchestration
// ---------------------------------------------------------------------------
extern "C" void kernel_launch(void* const* d_in, const int* in_sizes, int n_in,
                              void* d_out, int out_size, void* d_ws, size_t ws_size,
                              hipStream_t stream) {
    (void)in_sizes; (void)n_in; (void)out_size; (void)ws_size;
    const float* x     = (const float*)d_in[0];
    const float* W_ih1 = (const float*)d_in[1];
    const float* W_hh1 = (const float*)d_in[2];
    const float* b_ih1 = (const float*)d_in[3];
    const float* b_hh1 = (const float*)d_in[4];
    const float* W_ih2 = (const float*)d_in[5];
    const float* W_hh2 = (const float*)d_in[6];
    const float* b_ih2 = (const float*)d_in[7];
    const float* b_hh2 = (const float*)d_in[8];
    const float* W_p   = (const float*)d_in[9];
    const float* b_p   = (const float*)d_in[10];
    /* W_u */            const float* W_u = (const float*)d_in[11];
    const float* b_u   = (const float*)d_in[12];
    const float* W_e   = (const float*)d_in[13];
    const float* b_e   = (const float*)d_in[14];
    const float* W_o1  = (const float*)d_in[15];
    const float* b_o1  = (const float*)d_in[16];
    const float* W_o2  = (const float*)d_in[17];
    const float* b_o2  = (const float*)d_in[18];
    const float* h0    = (const float*)d_in[19];
    const float* c0    = (const float*)d_in[20];
    const float* h20   = (const float*)d_in[21];
    const float* c20   = (const float*)d_in[22];

    char* ws = (char*)d_ws;
    size_t off = 0;
    auto take = [&](size_t bytes) -> void* {
        void* p = ws + off;
        off = (off + bytes + 255) & ~(size_t)255;
        return p;
    };
    bf16_t* Wih1h  = (bf16_t*)take((size_t)G4H * Csz * 2);
    bf16_t* Whh1h  = (bf16_t*)take((size_t)G4H * Hsz * 2);
    bf16_t* Wcat2h = (bf16_t*)take((size_t)G4H * 512 * 2);
    bf16_t* Wuh    = (bf16_t*)take((size_t)128 * Hsz * 2);
    bf16_t* Wo1h   = (bf16_t*)take((size_t)512 * 1024 * 2);
    bf16_t* Wo2h   = (bf16_t*)take((size_t)256 * 512 * 2);
    float*  bias1  = (float*)take(G4H * 4);
    float*  bias2  = (float*)take(G4H * 4);
    float*  state  = (float*)take(4 * Bsz * Hsz * 4);
    bf16_t* xh     = (bf16_t*)take((size_t)ROWS * Csz * 2);
    float*  XG     = (float*)take((size_t)ROWS * G4H * 4);
    float*  hs     = (float*)take((size_t)ROWS * Hsz * 4);
    float*  cs     = (float*)take((size_t)ROWS * Hsz * 4);
    float*  outerh = (float*)take((size_t)NCH * Bsz * Hsz * 4);
    bf16_t* outhb  = (bf16_t*)take((size_t)NCH * Bsz * Hsz * 2);
    float*  ut     = (float*)take((size_t)NCH * Bsz * 128 * 4);
    bf16_t* flatb  = (bf16_t*)take((size_t)Bsz * 1024 * 2);

    const int T256 = 256;
    auto blks = [](int n) { return (n + 255) / 256; };

    // --- prep (cheap, once per call) ---
    k_cvt<<<blks(G4H * Csz), T256, 0, stream>>>(Wih1h, W_ih1, G4H * Csz);
    k_cvt<<<blks(G4H * Hsz), T256, 0, stream>>>(Whh1h, W_hh1, G4H * Hsz);
    k_wcat2<<<blks(G4H * 512), T256, 0, stream>>>(Wcat2h, W_ih2, W_hh2);
    k_cvt<<<blks(128 * Hsz), T256, 0, stream>>>(Wuh, W_u, 128 * Hsz);
    k_cvt<<<blks(512 * 1024), T256, 0, stream>>>(Wo1h, W_o1, 512 * 1024);
    k_cvt<<<blks(256 * 512), T256, 0, stream>>>(Wo2h, W_o2, 256 * 512);
    k_bias_sum<<<blks(G4H), T256, 0, stream>>>(bias1, b_ih1, b_hh1, G4H);
    k_bias_sum<<<blks(G4H), T256, 0, stream>>>(bias2, b_ih2, b_hh2, G4H);
    k_init_state<<<blks(Bsz * Hsz), T256, 0, stream>>>(state, h0, c0, h20, c20);

    // --- main loop over 80 chunks ---
    for (int ch = 0; ch < NCH; ++ch) {
        k_cvt_x<<<blks(ROWS * Csz), T256, 0, stream>>>(xh, x, ch * UPL);
        k_xgate<<<1280, T256, 0, stream>>>(xh, Wih1h, bias1, XG);   // 10240 tiles
        k_chunk_seq<<<1, 1024, CS_LDS_BYTES, stream>>>(
            XG, Whh1h, Wcat2h, bias2, W_p, b_p, state, hs, cs, outerh, outhb, ch);
    }

    // --- encoder + head ---
    k_enc_ut<<<160, T256, 0, stream>>>(outhb, Wuh, b_u, ut);        // 1280 tiles
    k_enc_att<<<1, 1024, 0, stream>>>(ut, W_e, b_e, outerh, flatb);
    k_mlp<<<1, 1024, 0, stream>>>(flatb, Wo1h, b_o1, Wo2h, b_o2, (float*)d_out);
}
